// VisionLanguageActionModel_69853348102805
// MI455X (gfx1250) — compile-verified, matrix operand order checked by reading, not run on hardware
//
#include <hip/hip_runtime.h>
#include <hip/hip_bf16.h>

// ---------------- problem constants ----------------
constexpr int B_  = 8;
constexpr int L_  = 1024;
constexpr int D_  = 768;
constexpr int DI_ = 1536;
constexpr int DS_ = 16;
constexpr int NL_ = 2;
constexpr int DC_ = 4;
constexpr int IP_ = 2 * DI_;        // 3072, in_proj output width
constexpr int M_  = B_ * L_;        // 8192 rows
constexpr float DECAY   = 0.9f;
constexpr float INSCALE = 0.1f;
constexpr float LN_EPS  = 1e-5f;

constexpr int CHUNK  = 128;         // scan chunk length
constexpr int NCHUNK = L_ / CHUNK;  // 8

typedef __attribute__((ext_vector_type(16))) _Float16 v16h;
typedef __attribute__((ext_vector_type(8)))  _Float16 v8h;
typedef __attribute__((ext_vector_type(8)))  float    v8f;
typedef __attribute__((ext_vector_type(4)))  float    v4f;
typedef __attribute__((ext_vector_type(4)))  int      v4i;

#define AS1 __attribute__((address_space(1)))
#define AS3 __attribute__((address_space(3)))

#if __has_builtin(__builtin_amdgcn_global_load_async_to_lds_b128)
#define HAVE_ASYNC_LDS 1
#else
#define HAVE_ASYNC_LDS 0
#endif

__device__ __forceinline__ void wait_asynccnt0() {
#if __has_builtin(__builtin_amdgcn_s_wait_asynccnt)
  __builtin_amdgcn_s_wait_asynccnt(0);
#else
  asm volatile("s_wait_asynccnt 0x0" ::: "memory");
#endif
}

__device__ __forceinline__ v8h zero8h() { v8h z = {}; return z; }
__device__ __forceinline__ v16h cat16(v8h lo, v8h hi) {
  return __builtin_shufflevector(lo, hi, 0,1,2,3,4,5,6,7,8,9,10,11,12,13,14,15);
}
// load 8 consecutive f32 (16B aligned) and convert to 8 f16
__device__ __forceinline__ v8h cvt8(const float* p) {
  v4f a = *(const v4f*)p;
  v4f b = *(const v4f*)(p + 4);
  v8h r;
  r[0]=(_Float16)a[0]; r[1]=(_Float16)a[1]; r[2]=(_Float16)a[2]; r[3]=(_Float16)a[3];
  r[4]=(_Float16)b[0]; r[5]=(_Float16)b[1]; r[6]=(_Float16)b[2]; r[7]=(_Float16)b[3];
  return r;
}
__device__ __forceinline__ float silu_f(float v) { return v / (1.0f + __expf(-v)); }

// ======================================================================
// Generic f32-in / f32-out GEMM via f16 WMMA, f32 accumulate.
// C[m,n] = (HAS_RESID ? resid[m,n] : 0) + sum_k A[m,k]*B[k,n]
// Workgroup tile 128x128, 8 waves, each wave: 16 rows x 128 cols.
// A tile staged fp32 (async global->LDS when available); B tile staged
// transposed f16.  Fragments use the documented 16-bit A layout:
// lane = row(col), VGPR v -> K pair (v>>2)*16 + half*8 + (v&3)*2.
// ======================================================================
#define A_STRIDE 40   // 32 + 8 pad floats: 160B rows, 16B aligned, banks spread
#define B_STRIDE 48   // halfs: 96B rows, 16B aligned

template <bool HAS_RESID>
__global__ __launch_bounds__(256)
void gemm_wmma_kernel(const float* __restrict__ A, const float* __restrict__ Bw,
                      const float* __restrict__ resid, float* __restrict__ C,
                      int M, int N, int K) {
  __shared__ float    Asf[128 * A_STRIDE];
  __shared__ _Float16 Bs [128 * B_STRIDE];

  const int t    = threadIdx.x;
  const int lane = t & 31;
  const int w    = t >> 5;          // wave 0..7
  const int m0   = blockIdx.y * 128;
  const int n0   = blockIdx.x * 128;
  const int mloc = lane & 15;
  const int half = lane >> 4;

  v8f acc0 = {0.f,0.f,0.f,0.f,0.f,0.f,0.f,0.f};
  v8f acc[8] = {acc0, acc0, acc0, acc0, acc0, acc0, acc0, acc0};

  // staging index precompute
  const int ar = t >> 3;            // 0..31 (A row group)
  const int ac = (t & 7) * 4;       // 0..28 (A col, one 16B chunk)
  const int bn = (t & 31) * 4;      // 0..124 (B col, 4 floats)
  const int bk = t >> 5;            // 0..7  (B row group)

  for (int k0 = 0; k0 < K; k0 += 32) {
    // ---- stage A tile: 128x32 f32, async DMA into LDS if available ----
    #pragma unroll
    for (int i = 0; i < 4; ++i) {
      const int row = ar + i * 32;
      const float* gsrc = A + (size_t)(m0 + row) * K + k0 + ac;
      float* ldst = &Asf[row * A_STRIDE + ac];
#if HAVE_ASYNC_LDS
      __builtin_amdgcn_global_load_async_to_lds_b128(
          (AS1 v4i*)gsrc, (AS3 v4i*)ldst, 0, 0);
#else
      *(v4f*)ldst = *(const v4f*)gsrc;
#endif
    }
    // ---- stage B tile transposed: B[k0+k][n0+n] -> Bs[n][k] (f16) ----
    #pragma unroll
    for (int i = 0; i < 4; ++i) {
      const int k = bk + i * 8;
      v4f bv = *(const v4f*)(Bw + (size_t)(k0 + k) * N + n0 + bn);
      Bs[(bn + 0) * B_STRIDE + k] = (_Float16)bv[0];
      Bs[(bn + 1) * B_STRIDE + k] = (_Float16)bv[1];
      Bs[(bn + 2) * B_STRIDE + k] = (_Float16)bv[2];
      Bs[(bn + 3) * B_STRIDE + k] = (_Float16)bv[3];
    }
#if HAVE_ASYNC_LDS
    wait_asynccnt0();
#endif
    __syncthreads();

    // ---- A fragment for this wave's 16-row slab (cvt f32->f16 here) ----
    const float* arow = &Asf[(w * 16 + mloc) * A_STRIDE];
    v16h afrag = cat16(cvt8(arow + half * 8), cvt8(arow + 16 + half * 8));

    #pragma unroll
    for (int j = 0; j < 8; ++j) {
      const _Float16* brow = &Bs[(j * 16 + mloc) * B_STRIDE];
      v8h blo = *(const v8h*)(brow + half * 8);
      v8h bhi = *(const v8h*)(brow + 16 + half * 8);
      v16h bfrag = cat16(blo, bhi);
      acc[j] = __builtin_amdgcn_wmma_f32_16x16x32_f16(
          false, afrag, false, bfrag, (short)0, acc[j], false, false);
    }
    __syncthreads();
  }

  // epilogue: VGPR r -> row half*8+r, col = lane&15 (branch-free)
  const size_t base = (size_t)(m0 + w * 16 + half * 8) * N + n0 + mloc;
  #pragma unroll
  for (int j = 0; j < 8; ++j) {
    #pragma unroll
    for (int r = 0; r < 8; ++r) {
      const size_t idx = base + (size_t)r * N + j * 16;
      float v = acc[j][r];
      if (HAS_RESID) v += resid[idx];
      C[idx] = v;
    }
  }
}

// ======================================================================
// Causal depthwise conv (DC=4) + SiLU on first DI lanes of xz.
// ======================================================================
__global__ __launch_bounds__(256)
void conv_silu_kernel(const float* __restrict__ xz, const float* __restrict__ cw,
                      const float* __restrict__ cb, float* __restrict__ xc) {
  const int tid = blockIdx.x * blockDim.x + threadIdx.x;
  const int c = tid % DI_;
  const int l = (tid / DI_) % L_;
  const int b = tid / (DI_ * L_);
  float acc = cb[c];
  #pragma unroll
  for (int k = 0; k < DC_; ++k) {
    const int lp = l + k - (DC_ - 1);
    if (lp >= 0)
      acc += cw[c * DC_ + k] * xz[((size_t)b * L_ + lp) * IP_ + c];
  }
  xc[((size_t)b * L_ + l) * DI_ + c] = silu_f(acc);
}

// ======================================================================
// u[l][b*16+s] = INSCALE * sum_c xc[b,l,c] * A_l[c,s]
// One wave per timestep l; M-tile = 16 (rows 0..7 = batch, 8..15 zero-pad).
// ======================================================================
__global__ __launch_bounds__(256)
void u_kernel(const float* __restrict__ xc, const float* __restrict__ Atl,
              float* __restrict__ u) {
  const int t    = threadIdx.x;
  const int lane = t & 31;
  const int w    = t >> 5;
  const int l    = blockIdx.x * 8 + w;
  const int m    = lane & 15;
  const int half = lane >> 4;

  v8f acc = {0.f,0.f,0.f,0.f,0.f,0.f,0.f,0.f};
  const float* ap = Atl + (size_t)l * (DI_ * DS_) + m;   // column n = lane&15

  for (int kk = 0; kk < DI_; kk += 32) {
    v16h af;
    if (m < 8) {
      const float* p = xc + ((size_t)m * L_ + l) * DI_ + kk;
      af = cat16(cvt8(p + half * 8), cvt8(p + 16 + half * 8));
    } else {
      af = cat16(zero8h(), zero8h());
    }
    const float* apk = ap + (size_t)kk * DS_;
    // prefetch next 32xDS slab of the streaming A tensor (HBM roofline term)
    __builtin_prefetch(apk + 32 * DS_, 0, 1);
    v16h bf;
    #pragma unroll
    for (int e = 0; e < 8; ++e) bf[e]     = (_Float16)apk[(half * 8 + e) * DS_];
    #pragma unroll
    for (int e = 0; e < 8; ++e) bf[8 + e] = (_Float16)apk[(16 + half * 8 + e) * DS_];

    acc = __builtin_amdgcn_wmma_f32_16x16x32_f16(
        false, af, false, bf, (short)0, acc, false, false);
  }

  if (half == 0) {   // rows 0..7 = batches
    #pragma unroll
    for (int r = 0; r < 8; ++r)
      u[(size_t)l * (B_ * DS_) + r * DS_ + m] = INSCALE * acc[r];
  }
}

// ======================================================================
// Chunked linear-decay scan: h_t = DECAY*h_{t-1} + u_t, 128 series.
// ======================================================================
__global__ void scan_pass1(const float* __restrict__ u, float* __restrict__ csum) {
  const int c = blockIdx.x, j = threadIdx.x;
  const float* p = u + (size_t)c * CHUNK * 128 + j;
  float h = 0.f;
  for (int i = 0; i < CHUNK; ++i) h = DECAY * h + p[(size_t)i * 128];
  csum[c * 128 + j] = h;
}
__global__ void scan_pass2(const float* __restrict__ csum, float* __restrict__ cin) {
  const int j = threadIdx.x;
  const float dpc = __powf(DECAY, (float)CHUNK);
  float hin = 0.f;
  for (int c = 0; c < NCHUNK; ++c) {
    cin[c * 128 + j] = hin;
    hin = dpc * hin + csum[c * 128 + j];
  }
}
__global__ void scan_pass3(const float* __restrict__ u, const float* __restrict__ cin,
                           float* __restrict__ hs) {
  const int c = blockIdx.x, j = threadIdx.x;
  const float* p = u  + (size_t)c * CHUNK * 128 + j;
  float*       q = hs + (size_t)c * CHUNK * 128 + j;
  float h = cin[c * 128 + j];
  for (int i = 0; i < CHUNK; ++i) { h = DECAY * h + p[(size_t)i * 128]; q[(size_t)i * 128] = h; }
}

// ======================================================================
// y[b,l,c] = (sum_s hs[l,b,s]*Bm_l[s,c]) * silu(z[b,l,c]),  z from xz[:,DI:]
// One wave per (l, 16-col tile). Single WMMA, K padded 16->32 with zeros.
// ======================================================================
__global__ __launch_bounds__(256)
void xssm_kernel(const float* __restrict__ hs, const float* __restrict__ Bml,
                 const float* __restrict__ xz, float* __restrict__ y) {
  const int t    = threadIdx.x;
  const int lane = t & 31;
  const int w    = t >> 5;
  const int gid  = blockIdx.x * 8 + w;
  const int l    = gid / (DI_ / 16);
  const int c0   = (gid % (DI_ / 16)) * 16;
  const int m    = lane & 15;
  const int half = lane >> 4;

  v8h lo = (m < 8) ? cvt8(hs + (size_t)l * (B_ * DS_) + m * DS_ + half * 8) : zero8h();
  v16h af = cat16(lo, zero8h());

  v16h bf;
  const float* bp = Bml + (size_t)l * (DS_ * DI_) + c0 + m;
  #pragma unroll
  for (int e = 0; e < 8; ++e) bf[e] = (_Float16)bp[(half * 8 + e) * DI_];
  #pragma unroll
  for (int e = 0; e < 8; ++e) bf[8 + e] = (_Float16)0.f;

  v8f acc = {0.f,0.f,0.f,0.f,0.f,0.f,0.f,0.f};
  acc = __builtin_amdgcn_wmma_f32_16x16x32_f16(
      false, af, false, bf, (short)0, acc, false, false);

  if (half == 0) {
    #pragma unroll
    for (int r = 0; r < 8; ++r) {
      const size_t row = (size_t)r * L_ + l;          // batch r, time l
      const float z = xz[row * IP_ + DI_ + c0 + m];
      y[row * DI_ + c0 + m] = acc[r] * silu_f(z);
    }
  }
}

// ======================================================================
// Final LayerNorm over D=768, two-pass, one 256-thread block per row.
// ======================================================================
__global__ __launch_bounds__(256)
void ln_kernel(const float* __restrict__ x, const float* __restrict__ gamma,
               const float* __restrict__ beta, float* __restrict__ out) {
  __shared__ float red[256];
  const int row = blockIdx.x, t = threadIdx.x;
  const float* xr = x + (size_t)row * D_;
  float v0 = xr[t], v1 = xr[t + 256], v2 = xr[t + 512];
  red[t] = v0 + v1 + v2;
  __syncthreads();
  for (int o = 128; o > 0; o >>= 1) { if (t < o) red[t] += red[t + o]; __syncthreads(); }
  const float mu = red[0] * (1.0f / D_);
  __syncthreads();
  const float d0 = v0 - mu, d1 = v1 - mu, d2 = v2 - mu;
  red[t] = d0 * d0 + d1 * d1 + d2 * d2;
  __syncthreads();
  for (int o = 128; o > 0; o >>= 1) { if (t < o) red[t] += red[t + o]; __syncthreads(); }
  const float rs = rsqrtf(red[0] * (1.0f / D_) + LN_EPS);
  float* orow = out + (size_t)row * D_;
  orow[t]       = d0 * rs * gamma[t]       + beta[t];
  orow[t + 256] = d1 * rs * gamma[t + 256] + beta[t + 256];
  orow[t + 512] = d2 * rs * gamma[t + 512] + beta[t + 512];
}

// ======================================================================
extern "C" void kernel_launch(void* const* d_in, const int* in_sizes, int n_in,
                              void* d_out, int out_size, void* d_ws, size_t ws_size,
                              hipStream_t stream) {
  const float* x     = (const float*)d_in[0];
  const float* Wi    = (const float*)d_in[1];  // [NL][D][2*DI]
  const float* cw    = (const float*)d_in[2];  // [NL][DI][DC]
  const float* cb    = (const float*)d_in[3];  // [NL][DI]
  const float* Wo    = (const float*)d_in[4];  // [NL][DI][D]
  const float* gamma = (const float*)d_in[5];
  const float* beta  = (const float*)d_in[6];
  const float* A     = (const float*)d_in[7];  // [NL][L][DI][DS]
  const float* Bm    = (const float*)d_in[8];  // [NL][L][DS][DI]

  float* ws = (float*)d_ws;
  float* xz   = ws;                       ws += (size_t)M_ * IP_;   // 25.2M
  float* xc   = ws;                       ws += (size_t)M_ * DI_;   // 12.6M (reused as y)
  float* xcur = ws;                       ws += (size_t)M_ * D_;    //  6.3M
  float* u    = ws;                       ws += (size_t)L_ * B_ * DS_;
  float* hs   = ws;                       ws += (size_t)L_ * B_ * DS_;
  float* csum = ws;                       ws += NCHUNK * 128;
  float* cin  = ws;                       ws += NCHUNK * 128;

  const float* xprev = x;
  for (int layer = 0; layer < NL_; ++layer) {
    // in_proj: xz = xprev @ Wi[layer]   (M=8192, N=3072, K=768)
    gemm_wmma_kernel<false><<<dim3(IP_ / 128, M_ / 128), 256, 0, stream>>>(
        xprev, Wi + (size_t)layer * D_ * IP_, nullptr, xz, M_, IP_, D_);

    conv_silu_kernel<<<(size_t)M_ * DI_ / 256, 256, 0, stream>>>(
        xz, cw + (size_t)layer * DI_ * DC_, cb + (size_t)layer * DI_, xc);

    u_kernel<<<L_ / 8, 256, 0, stream>>>(
        xc, A + (size_t)layer * L_ * DI_ * DS_, u);

    scan_pass1<<<NCHUNK, 128, 0, stream>>>(u, csum);
    scan_pass2<<<1, 128, 0, stream>>>(csum, cin);
    scan_pass3<<<NCHUNK, 128, 0, stream>>>(u, cin, hs);

    // y (into xc buffer): gated state projection
    xssm_kernel<<<(L_ * (DI_ / 16)) / 8, 256, 0, stream>>>(
        hs, Bm + (size_t)layer * L_ * DS_ * DI_, xz, xc);

    // out_proj with residual: xcur = xprev + y @ Wo[layer]
    gemm_wmma_kernel<true><<<dim3(D_ / 128, M_ / 128), 256, 0, stream>>>(
        xc, Wo + (size_t)layer * DI_ * D_, xprev, xcur, M_, D_, DI_);

    xprev = xcur;
  }

  ln_kernel<<<M_, 256, 0, stream>>>(xcur, gamma, beta, (float*)d_out);
}
